// PointMLPModel_57853209477796
// MI455X (gfx1250) — matrix operand order repaired and checked by reading, not compiled, not run on hardware
//
#include <hip/hip_runtime.h>
#include <hip/hip_bf16.h>

// ---------------------------------------------------------------------------
// PointMLP forward for MI455X (gfx1250, wave32).
// GEMMs: v_wmma_f32_16x16x32_f16 (f16 inputs, f32 accumulate), 2x2 register
// blocking per wave (32x32 output block) for 2x operand reuse.
// Activation layout: N-major (row = sample index, col = channel).
// ---------------------------------------------------------------------------

typedef _Float16 half_t;
typedef _Float16 h8   __attribute__((ext_vector_type(8)));
typedef _Float16 v16h __attribute__((ext_vector_type(16)));
typedef float    v8f  __attribute__((ext_vector_type(8)));
typedef float    f4   __attribute__((ext_vector_type(4)));

__device__ inline v8f wmma_f16(const v16h& a, const v16h& b, const v8f& c) {
    return __builtin_amdgcn_wmma_f32_16x16x32_f16(false, a, false, b, (short)0, c,
                                                  false, false);
}

// ---------------- WMMA GEMM (1x1 tile / wave, generic fallback) -------------
// Y(N x M) = X(N x Kc) * W(M x Kc)^T ; per-lane addressing per ISA 7.12.2.
__global__ __launch_bounds__(256) void gemm_h16_kernel(
    const half_t* __restrict__ W, const half_t* __restrict__ X,
    float* __restrict__ Y, int M, int Kc, int N)
{
    int wave   = (blockIdx.x << 3) + (threadIdx.x >> 5);
    int ntiles = N >> 4;
    int total  = (M >> 4) * ntiles;
    if (wave >= total) return;                 // wave-uniform: EXEC stays all-1s
    int mt = wave / ntiles, nt = wave % ntiles;
    int lane = threadIdx.x & 31;
    int lh = lane & 15, hh = lane >> 4;

    const half_t* wrow = W + (size_t)(mt * 16 + lh) * Kc + hh * 8;
    const half_t* xrow = X + (size_t)(nt * 16 + lh) * Kc + hh * 8;

    v8f c = {};
    for (int k0 = 0; k0 < Kc; k0 += 32) {
        union { v16h v; h8 h[2]; } a, b;
        a.h[0] = *(const h8*)(wrow + k0);
        a.h[1] = *(const h8*)(wrow + k0 + 16);
        b.h[0] = *(const h8*)(xrow + k0);
        b.h[1] = *(const h8*)(xrow + k0 + 16);
        c = wmma_f16(a.v, b.v, c);
    }
    float* yp = Y + (size_t)(nt * 16 + lh) * M + mt * 16 + hh * 8;
    union { v8f v; f4 f[2]; } u; u.v = c;
    *(f4*)yp       = u.f[0];
    *(f4*)(yp + 4) = u.f[1];
}

// ---------------- WMMA GEMM (2x2 tiles / wave) ------------------------------
// Requires M%32==0, N%32==0. Each wave computes a 32x32 output block:
// 4 accumulators, 4 operand loads -> 4 WMMAs per K-step (16 flops/byte).
__global__ __launch_bounds__(256) void gemm_h16_2x2_kernel(
    const half_t* __restrict__ W, const half_t* __restrict__ X,
    float* __restrict__ Y, int M, int Kc, int N)
{
    int wave  = (blockIdx.x << 3) + (threadIdx.x >> 5);
    int nb    = N >> 5;
    int total = (M >> 5) * nb;
    if (wave >= total) return;
    int mb = wave / nb, nbi = wave % nb;
    int lane = threadIdx.x & 31;
    int lh = lane & 15, hh = lane >> 4;

    const half_t* wrow0 = W + (size_t)(mb * 32 + lh) * Kc + hh * 8;
    const half_t* wrow1 = wrow0 + (size_t)16 * Kc;
    const half_t* xrow0 = X + (size_t)(nbi * 32 + lh) * Kc + hh * 8;
    const half_t* xrow1 = xrow0 + (size_t)16 * Kc;

    v8f c00 = {}, c01 = {}, c10 = {}, c11 = {};
    for (int k0 = 0; k0 < Kc; k0 += 32) {
        union { v16h v; h8 h[2]; } a0, a1, b0, b1;
        a0.h[0] = *(const h8*)(wrow0 + k0);
        a0.h[1] = *(const h8*)(wrow0 + k0 + 16);
        a1.h[0] = *(const h8*)(wrow1 + k0);
        a1.h[1] = *(const h8*)(wrow1 + k0 + 16);
        b0.h[0] = *(const h8*)(xrow0 + k0);
        b0.h[1] = *(const h8*)(xrow0 + k0 + 16);
        b1.h[0] = *(const h8*)(xrow1 + k0);
        b1.h[1] = *(const h8*)(xrow1 + k0 + 16);
        c00 = wmma_f16(a0.v, b0.v, c00);
        c01 = wmma_f16(a0.v, b1.v, c01);
        c10 = wmma_f16(a1.v, b0.v, c10);
        c11 = wmma_f16(a1.v, b1.v, c11);
    }
    // tile (mi, ni): rows = nbi*32 + ni*16 + lh ; cols = mb*32 + mi*16 + hh*8
    {
        union { v8f v; f4 f[2]; } u;
        float* y00 = Y + (size_t)(nbi * 32 + lh) * M + mb * 32 + hh * 8;
        u.v = c00; *(f4*)y00 = u.f[0]; *(f4*)(y00 + 4) = u.f[1];
        float* y10 = y00 + 16;
        u.v = c10; *(f4*)y10 = u.f[0]; *(f4*)(y10 + 4) = u.f[1];
        float* y01 = Y + (size_t)(nbi * 32 + 16 + lh) * M + mb * 32 + hh * 8;
        u.v = c01; *(f4*)y01 = u.f[0]; *(f4*)(y01 + 4) = u.f[1];
        float* y11 = y01 + 16;
        u.v = c11; *(f4*)y11 = u.f[0]; *(f4*)(y11 + 4) = u.f[1];
    }
}

// ---------------- small utility kernels ------------------------------------
__global__ void zero_kernel(float* __restrict__ p, int n) {
    int i = blockIdx.x * 256 + threadIdx.x;
    if (i < n) p[i] = 0.0f;
}

__global__ void f2h_kernel(const float* __restrict__ s, half_t* __restrict__ d, size_t n) {
    size_t i = (size_t)blockIdx.x * 256 + threadIdx.x;
    if (i < n) d[i] = (half_t)s[i];
}

__global__ void padw3_kernel(const float* __restrict__ w3, half_t* __restrict__ o) {
    int i = blockIdx.x * 256 + threadIdx.x;   // 48*256
    if (i >= 48 * 256) return;
    int r = i / 256, c = i % 256;
    o[i] = (r < 40) ? (half_t)w3[r * 256 + c] : (half_t)0.0f;
}

// ---------------- BN (training stats over all rows, per channel) -----------
__global__ void bnstat_kernel(const float* __restrict__ Y,
                              float* __restrict__ gsum, float* __restrict__ gsq,
                              int Nr, int M)
{
    int m = blockIdx.y * 256 + threadIdx.x;
    if (m >= M) return;
    float s = 0.f, ss = 0.f;
    for (int n = blockIdx.x; n < Nr; n += gridDim.x) {
        float v = Y[(size_t)n * M + m];
        s += v; ss += v * v;
    }
    atomicAdd(&gsum[m], s);
    atomicAdd(&gsq[m], ss);
}

__global__ void bnfin_kernel(const float* __restrict__ gsum, const float* __restrict__ gsq,
                             const float* __restrict__ g, const float* __restrict__ b,
                             float* __restrict__ scale, float* __restrict__ shift,
                             int M, float invN)
{
    int m = blockIdx.x * 256 + threadIdx.x;
    if (m >= M) return;
    float mean = gsum[m] * invN;
    float var  = gsq[m] * invN - mean * mean;
    float s    = g[m] * rsqrtf(var + 1e-5f);
    scale[m] = s;
    shift[m] = b[m] - mean * s;
}

// v = scale*y + shift ; (+res) ; (relu) ; emit f32 and/or f16
__global__ void bnapply_kernel(const float* __restrict__ Yin,
                               const float* __restrict__ scale, const float* __restrict__ shift,
                               const float* __restrict__ res,
                               float* __restrict__ o32, half_t* __restrict__ o16,
                               size_t total, int M, int relu)
{
    size_t i = (size_t)blockIdx.x * 256 + threadIdx.x;
    if (i >= total) return;
    int m = (int)(i % M);
    float v = scale[m] * Yin[i] + shift[m];
    if (res)  v += res[i];
    if (relu) v = fmaxf(v, 0.f);
    if (o32) o32[i] = v;
    if (o16) o16[i] = (half_t)v;
}

// ---------------- embed / xyz ----------------------------------------------
__global__ void embed_kernel(const float* __restrict__ x, const float* __restrict__ W,
                             float* __restrict__ Y, int Np)
{
    int i = blockIdx.x * 256 + threadIdx.x;   // over B*Np
    if (i >= 16 * Np) return;
    int b = i / Np, l = i % Np;
    const float* xb = x + (size_t)b * 3 * Np + l;
    float x0 = xb[0], x1 = xb[Np], x2 = xb[2 * Np];
    float* y = Y + (size_t)i * 64;
    for (int o = 0; o < 64; ++o)
        y[o] = W[o * 3] * x0 + W[o * 3 + 1] * x1 + W[o * 3 + 2] * x2;
}

__global__ void xyz_kernel(const float* __restrict__ x, float* __restrict__ xyz, int Np) {
    int i = blockIdx.x * 256 + threadIdx.x;
    if (i >= 16 * Np) return;
    int b = i / Np, l = i % Np;
    const float* xb = x + (size_t)b * 3 * Np + l;
    xyz[(size_t)i * 3 + 0] = xb[0];
    xyz[(size_t)i * 3 + 1] = xb[Np];
    xyz[(size_t)i * 3 + 2] = xb[2 * Np];
}

// ---------------- farthest point sampling (sequential, 1 block / batch) ----
__global__ void fps_kernel(const float* __restrict__ xyz, int* __restrict__ fidx,
                           float* __restrict__ dist, int N, int S)
{
    int b = blockIdx.x;
    const float* px = xyz + (size_t)b * N * 3;
    float* d = dist + (size_t)b * N;
    __shared__ float sv[256]; __shared__ int si[256]; __shared__ int sfar;
    int t = threadIdx.x;
    for (int n = t; n < N; n += 256) d[n] = 1e10f;
    if (t == 0) sfar = 0;
    __syncthreads();
    for (int it = 0; it < S; ++it) {
        int far = sfar;
        if (t == 0) fidx[b * S + it] = far;
        float cx = px[far * 3], cy = px[far * 3 + 1], cz = px[far * 3 + 2];
        float bv = -1.f; int bi = 0;
        for (int n = t; n < N; n += 256) {
            float dx = px[n * 3] - cx, dy = px[n * 3 + 1] - cy, dz = px[n * 3 + 2] - cz;
            float dd = dx * dx + dy * dy + dz * dz;
            float od = d[n];
            float nd = od < dd ? od : dd;
            d[n] = nd;
            if (nd > bv) { bv = nd; bi = n; }   // strict > => first max in thread slice
        }
        sv[t] = bv; si[t] = bi;
        __syncthreads();
        for (int s = 128; s > 0; s >>= 1) {
            if (t < s) {
                float v2 = sv[t + s]; int i2 = si[t + s];
                if (v2 > sv[t] || (v2 == sv[t] && i2 < si[t])) { sv[t] = v2; si[t] = i2; }
            }
            __syncthreads();
        }
        if (t == 0) sfar = si[0];
        __syncthreads();
    }
}

__global__ void gatherxyz_kernel(const float* __restrict__ xyz, const int* __restrict__ fidx,
                                 float* __restrict__ nxyz, int N, int S)
{
    int i = blockIdx.x * 256 + threadIdx.x;   // over B*S
    if (i >= 16 * S) return;
    int b = i / S;
    int f = fidx[i];
    const float* p = xyz + ((size_t)b * N + f) * 3;
    nxyz[(size_t)i * 3 + 0] = p[0];
    nxyz[(size_t)i * 3 + 1] = p[1];
    nxyz[(size_t)i * 3 + 2] = p[2];
}

// ---------------- kNN: 24 smallest of d = |q|^2 + |p|^2 - 2 q.p ------------
__global__ void knn_kernel(const float* __restrict__ xyz, const float* __restrict__ nxyz,
                           int* __restrict__ kidx, int N, int S, int Kk)
{
    int s = blockIdx.x, b = blockIdx.y;
    const float* px = xyz + (size_t)b * N * 3;
    const float* q  = nxyz + ((size_t)b * S + s) * 3;
    extern __shared__ float dsh[];            // N floats
    __shared__ float sv[256]; __shared__ int si[256];
    int t = threadIdx.x;
    float qx = q[0], qy = q[1], qz = q[2];
    float q2 = qx * qx + qy * qy + qz * qz;
    for (int n = t; n < N; n += 256) {
        float x = px[n * 3], y = px[n * 3 + 1], z = px[n * 3 + 2];
        dsh[n] = q2 + (x * x + y * y + z * z) - 2.f * (qx * x + qy * y + qz * z);
    }
    __syncthreads();
    for (int j = 0; j < Kk; ++j) {
        float bv = 3.4e38f; int bi = 0;
        for (int n = t; n < N; n += 256) { float v = dsh[n]; if (v < bv) { bv = v; bi = n; } }
        sv[t] = bv; si[t] = bi;
        __syncthreads();
        for (int r = 128; r > 0; r >>= 1) {
            if (t < r) {
                if (sv[t + r] < sv[t] || (sv[t + r] == sv[t] && si[t + r] < si[t])) {
                    sv[t] = sv[t + r]; si[t] = si[t + r];
                }
            }
            __syncthreads();
        }
        if (t == 0) { kidx[((size_t)b * S + s) * Kk + j] = si[0]; dsh[si[0]] = 3.4e38f; }
        __syncthreads();
    }
}

// ---------------- per-batch std of diff (ddof=1) ---------------------------
__global__ void diffstat_kernel(const float* __restrict__ h, const int* __restrict__ kidx,
                                const int* __restrict__ fidx,
                                float* __restrict__ bsum, float* __restrict__ bsq,
                                int N, int S, int Kk, int C)
{
    int b = blockIdx.y;
    long total = (long)S * Kk * C;
    float s = 0.f, ss = 0.f;
    for (long i = (long)blockIdx.x * 256 + threadIdx.x; i < total; i += (long)gridDim.x * 256) {
        int c = (int)(i % C);
        long sk = i / C;
        int k = (int)(sk % Kk);
        int sd = (int)(sk / Kk);
        int pi = kidx[((size_t)b * S + sd) * Kk + k];
        int fi = fidx[b * S + sd];
        float diff = h[((size_t)b * N + pi) * C + c] - h[((size_t)b * N + fi) * C + c];
        s += diff; ss += diff * diff;
    }
    __shared__ float r1[256], r2[256];
    r1[threadIdx.x] = s; r2[threadIdx.x] = ss;
    __syncthreads();
    for (int r = 128; r > 0; r >>= 1) {
        if (threadIdx.x < r) { r1[threadIdx.x] += r1[threadIdx.x + r]; r2[threadIdx.x] += r2[threadIdx.x + r]; }
        __syncthreads();
    }
    if (threadIdx.x == 0) { atomicAdd(&bsum[b], r1[0]); atomicAdd(&bsq[b], r2[0]); }
}

__global__ void stdfin_kernel(const float* __restrict__ bsum, const float* __restrict__ bsq,
                              float* __restrict__ stdv, float n)
{
    int b = threadIdx.x;
    if (b >= 16) return;
    float s = bsum[b], ss = bsq[b];
    float var = (ss - s * s / n) / (n - 1.f);
    stdv[b] = sqrtf(fmaxf(var, 0.f));
}

// ---------------- grouping: write [g | rep] as f16 (rows = (b,s,k)) --------
__global__ void group_kernel(const float* __restrict__ h, const int* __restrict__ kidx,
                             const int* __restrict__ fidx,
                             const float* __restrict__ alpha, const float* __restrict__ beta,
                             const float* __restrict__ stdv,
                             half_t* __restrict__ X, size_t total, int N, int S, int Kk, int C)
{
    size_t i = (size_t)blockIdx.x * 256 + threadIdx.x;   // over B*S*Kk*C
    if (i >= total) return;
    int c = (int)(i % C);
    size_t r = i / C;
    int k = (int)(r % Kk);
    size_t bs = r / Kk;
    int b = (int)(bs / S);
    int pi = kidx[bs * Kk + k];
    int fi = fidx[bs];
    float mv = h[((size_t)b * N + fi) * C + c];
    float diff = h[((size_t)b * N + pi) * C + c] - mv;
    float g = alpha[c] * (diff / (stdv[b] + 1e-5f)) + beta[c];
    size_t row = bs * Kk + k;
    X[row * (size_t)(2 * C) + c]     = (half_t)g;
    X[row * (size_t)(2 * C) + C + c] = (half_t)mv;
}

// ---------------- max over Kk consecutive rows -----------------------------
__global__ void maxk_kernel(const float* __restrict__ P, float* __restrict__ o32,
                            half_t* __restrict__ o16, int BS, int Kk, int C)
{
    size_t i = (size_t)blockIdx.x * 256 + threadIdx.x;   // over BS*C
    if (i >= (size_t)BS * C) return;
    int c = (int)(i % C);
    size_t bs = i / C;
    const float* p = P + bs * (size_t)Kk * C + c;
    float v = p[0];
    for (int k = 1; k < Kk; ++k) v = fmaxf(v, p[(size_t)k * C]);
    if (o32) o32[i] = v;
    if (o16) o16[i] = (half_t)v;
}

// ---------------- final bias + copy to d_out -------------------------------
__global__ void out_kernel(const float* __restrict__ Y, const float* __restrict__ b3,
                           float* __restrict__ out)
{
    int i = blockIdx.x * 256 + threadIdx.x;   // 16*40
    if (i >= 16 * 40) return;
    int b = i / 40, j = i % 40;
    out[i] = Y[b * 48 + j] + b3[j];
}

// ===========================================================================
extern "C" void kernel_launch(void* const* d_in, const int* in_sizes, int n_in,
                              void* d_out, int out_size, void* d_ws, size_t ws_size,
                              hipStream_t stream)
{
    (void)in_sizes; (void)n_in; (void)out_size; (void)ws_size;
    auto F = [&](int i) { return (const float*)d_in[i]; };

    // Input leaf order: d_in[0] = x; then params via jax-style flatten
    // (dict keys sorted at every level, lists in order):
    // cls{b1,b2,b3,be1,be2,g1,g2,w1,w2,w3}, embed{b,g,w},
    // stages[i]{alpha,beta,pos[0..1]{b1,b2,g1,g2,w1,w2},
    //           pre.blocks[0..1]{...}, transfer_b, transfer_g, transfer_w}
    const float* x       = F(0);
    const int base = 1;
    const float* cls_b3  = F(base + 2);
    const float* cls_be1 = F(base + 3);
    const float* cls_be2 = F(base + 4);
    const float* cls_g1  = F(base + 5);
    const float* cls_g2  = F(base + 6);
    const float* cls_w1  = F(base + 7);
    const float* cls_w2  = F(base + 8);
    const float* cls_w3  = F(base + 9);
    const float* emb_b   = F(base + 10);
    const float* emb_g   = F(base + 11);
    const float* emb_w   = F(base + 12);

    struct Blk { const float *b1, *b2, *g1, *g2, *w1, *w2; half_t *w1h, *w2h; };
    struct Stage { const float *alpha, *beta, *tb, *tg, *tw; Blk pos[2], pre[2]; half_t* twh; int out; };
    Stage st[4];
    const int outs[4] = {128, 256, 512, 1024};
    for (int i = 0; i < 4; ++i) {
        int s0 = base + 13 + 29 * i;
        st[i].alpha = F(s0); st[i].beta = F(s0 + 1);
        auto rb = [&](int o) { Blk b; b.b1 = F(o); b.b2 = F(o + 1); b.g1 = F(o + 2);
                               b.g2 = F(o + 3); b.w1 = F(o + 4); b.w2 = F(o + 5);
                               b.w1h = b.w2h = nullptr; return b; };
        st[i].pos[0] = rb(s0 + 2);  st[i].pos[1] = rb(s0 + 8);
        st[i].pre[0] = rb(s0 + 14); st[i].pre[1] = rb(s0 + 20);
        st[i].tb = F(s0 + 26); st[i].tg = F(s0 + 27); st[i].tw = F(s0 + 28);
        st[i].out = outs[i];
    }

    // ---------------- workspace layout (needs ~340 MB) ---------------------
    char* wsp = (char*)d_ws; size_t off = 0;
    auto alloc = [&](size_t bytes) { void* p = wsp + off; off = (off + bytes + 255) & ~(size_t)255; return p; };
    const size_t ACT = 25165824;                  // 16*512*24*128 max activation elems
    float*  P     = (float*) alloc(ACT * 4);
    float*  R     = (float*) alloc(ACT * 4);
    half_t* H0    = (half_t*)alloc(ACT * 2);
    half_t* H1    = (half_t*)alloc(ACT * 2);
    float*  hbuf  = (float*) alloc((size_t)1048576 * 4);     // point features (B*N x C)
    float*  xyzA  = (float*) alloc((size_t)16 * 1024 * 3 * 4);
    float*  xyzB  = (float*) alloc((size_t)16 * 1024 * 3 * 4);
    int*    fidx  = (int*)   alloc((size_t)16 * 512 * 4);
    int*    kidx  = (int*)   alloc((size_t)16 * 512 * 24 * 4);
    float*  fdist = (float*) alloc((size_t)16 * 1024 * 4);
    float*  gsum  = (float*) alloc(1024 * 4);
    float*  gsq   = (float*) alloc(1024 * 4);
    float*  scale = (float*) alloc(1024 * 4);
    float*  shift = (float*) alloc(1024 * 4);
    float*  bsum  = (float*) alloc(16 * 4);
    float*  bsq   = (float*) alloc(16 * 4);
    float*  stdv  = (float*) alloc(16 * 4);
    float*  feat32= (float*) alloc((size_t)16 * 1024 * 4);
    half_t* featH = (half_t*)alloc((size_t)16 * 1024 * 2);
    half_t* hpool = (half_t*)alloc((size_t)13300000 * 2);

    // ---------------- f16 weight conversion (per call; deterministic) ------
    half_t* hp = hpool;
    auto take = [&](size_t n) { half_t* p = hp; hp += n; return p; };
    auto cvt  = [&](const float* s, half_t* d, size_t n) {
        f2h_kernel<<<(unsigned)((n + 255) / 256), 256, 0, stream>>>(s, d, n);
    };
    for (int i = 0; i < 4; ++i) {
        size_t nn = (size_t)outs[i] * outs[i];
        st[i].twh = take(nn); cvt(st[i].tw, st[i].twh, nn);
        for (int j = 0; j < 2; ++j) {
            st[i].pre[j].w1h = take(nn); cvt(st[i].pre[j].w1, st[i].pre[j].w1h, nn);
            st[i].pre[j].w2h = take(nn); cvt(st[i].pre[j].w2, st[i].pre[j].w2h, nn);
            st[i].pos[j].w1h = take(nn); cvt(st[i].pos[j].w1, st[i].pos[j].w1h, nn);
            st[i].pos[j].w2h = take(nn); cvt(st[i].pos[j].w2, st[i].pos[j].w2h, nn);
        }
    }
    half_t* w1c = take((size_t)512 * 1024); cvt(cls_w1, w1c, (size_t)512 * 1024);
    half_t* w2c = take((size_t)256 * 512);  cvt(cls_w2, w2c, (size_t)256 * 512);
    half_t* w3p = take((size_t)48 * 256);
    padw3_kernel<<<48, 256, 0, stream>>>(cls_w3, w3p);

    // ---------------- helpers ----------------------------------------------
    auto gemm = [&](const half_t* Wm, const half_t* Xm, float* Ym, int M, int Kc, int N) {
        if ((M % 32 == 0) && (N % 32 == 0)) {
            int waves = (M >> 5) * (N >> 5);
            gemm_h16_2x2_kernel<<<(unsigned)((waves + 7) / 8), 256, 0, stream>>>(
                Wm, Xm, Ym, M, Kc, N);
        } else {
            int waves = (M >> 4) * (N >> 4);
            gemm_h16_kernel<<<(unsigned)((waves + 7) / 8), 256, 0, stream>>>(
                Wm, Xm, Ym, M, Kc, N);
        }
    };
    auto bnrun = [&](float* Yr, long Nr, int M, const float* gg, const float* bb,
                     const float* res, float* o32, half_t* o16, int relu) {
        zero_kernel<<<4, 256, 0, stream>>>(gsum, 1024);
        zero_kernel<<<4, 256, 0, stream>>>(gsq, 1024);
        int gx = (int)(Nr < 256 ? Nr : 256);
        dim3 gs(gx, (M + 255) / 256);
        bnstat_kernel<<<gs, 256, 0, stream>>>(Yr, gsum, gsq, (int)Nr, M);
        bnfin_kernel<<<(M + 255) / 256, 256, 0, stream>>>(gsum, gsq, gg, bb, scale, shift, M,
                                                          1.0f / (float)Nr);
        size_t tot = (size_t)Nr * M;
        bnapply_kernel<<<(unsigned)((tot + 255) / 256), 256, 0, stream>>>(
            Yr, scale, shift, res, o32, o16, tot, M, relu);
    };

    // ---------------- embed + xyz ------------------------------------------
    embed_kernel<<<64, 256, 0, stream>>>(x, emb_w, P, 1024);
    bnrun(P, 16384, 64, emb_g, emb_b, nullptr, hbuf, nullptr, 1);
    xyz_kernel<<<64, 256, 0, stream>>>(x, xyzA, 1024);

    float* xyzCur = xyzA; float* xyzNew = xyzB;
    int Ncur = 1024, Ccur = 64;

    // ---------------- stages ------------------------------------------------
    for (int i = 0; i < 4; ++i) {
        int S = Ncur / 2;
        int out = st[i].out;
        long Ntot = (long)16 * S * 24;

        fps_kernel<<<16, 256, 0, stream>>>(xyzCur, fidx, fdist, Ncur, S);
        gatherxyz_kernel<<<(16 * S + 255) / 256, 256, 0, stream>>>(xyzCur, fidx, xyzNew, Ncur, S);
        knn_kernel<<<dim3(S, 16), 256, (size_t)Ncur * 4, stream>>>(xyzCur, xyzNew, kidx, Ncur, S, 24);

        zero_kernel<<<1, 64, 0, stream>>>(bsum, 16);
        zero_kernel<<<1, 64, 0, stream>>>(bsq, 16);
        diffstat_kernel<<<dim3(64, 16), 256, 0, stream>>>(hbuf, kidx, fidx, bsum, bsq, Ncur, S, 24, Ccur);
        stdfin_kernel<<<1, 32, 0, stream>>>(bsum, bsq, stdv, (float)((long)S * 24 * Ccur));

        size_t gtot = (size_t)16 * S * 24 * Ccur;
        group_kernel<<<(unsigned)((gtot + 255) / 256), 256, 0, stream>>>(
            hbuf, kidx, fidx, st[i].alpha, st[i].beta, stdv, H0, gtot, Ncur, S, 24, Ccur);

        // transfer conv + BN + ReLU
        gemm(st[i].twh, H0, P, out, 2 * Ccur, (int)Ntot);
        bnrun(P, Ntot, out, st[i].tg, st[i].tb, nullptr, P, H1, 1);

        // pre res-blocks over (Ntot x out)
        float* cur = P; float* fb = R; half_t* hc = H1; half_t* hf = H0;
        for (int j = 0; j < 2; ++j) {
            Blk& B = st[i].pre[j];
            gemm(B.w1h, hc, fb, out, out, (int)Ntot);
            bnrun(fb, Ntot, out, B.g1, B.b1, nullptr, nullptr, hf, 1);
            gemm(B.w2h, hf, fb, out, out, (int)Ntot);
            bnrun(fb, Ntot, out, B.g2, B.b2, cur, fb, hc, 1);
            float* t = cur; cur = fb; fb = t;
        }

        // max over K -> (B*S x out)
        int N2 = 16 * S;
        maxk_kernel<<<(unsigned)(((size_t)N2 * out + 255) / 256), 256, 0, stream>>>(
            cur, fb, hf, N2, 24, out);

        // pos res-blocks over (N2 x out)
        float* a32 = fb; float* fb2 = cur; half_t* ha = hf; half_t* hb = hc;
        for (int j = 0; j < 2; ++j) {
            Blk& B = st[i].pos[j];
            gemm(B.w1h, ha, fb2, out, out, N2);
            bnrun(fb2, N2, out, B.g1, B.b1, nullptr, nullptr, hb, 1);
            gemm(B.w2h, hb, fb2, out, out, N2);
            bnrun(fb2, N2, out, B.g2, B.b2, a32, fb2, ha, 1);
            float* t = a32; a32 = fb2; fb2 = t;
        }

        hipMemcpyAsync(hbuf, a32, (size_t)N2 * out * 4, hipMemcpyDeviceToDevice, stream);
        float* tx = xyzCur; xyzCur = xyzNew; xyzNew = tx;
        Ncur = S; Ccur = out;
    }

    // ---------------- classifier -------------------------------------------
    // feat = max over S (=64) -> (16 x 1024)
    maxk_kernel<<<64, 256, 0, stream>>>(hbuf, feat32, featH, 16, 64, 1024);
    // fc1: bias b1 cancels under batch-BN mean subtraction (exact)
    gemm(w1c, featH, P, 512, 1024, 16);
    bnrun(P, 16, 512, cls_g1, cls_be1, nullptr, nullptr, H0, 1);
    gemm(w2c, H0, P, 256, 512, 16);
    bnrun(P, 16, 256, cls_g2, cls_be2, nullptr, nullptr, H1, 1);
    gemm(w3p, H1, P, 48, 256, 16);
    out_kernel<<<3, 256, 0, stream>>>(P, cls_b3, (float*)d_out);
}
